// AttentionQKV_2757369004072
// MI455X (gfx1250) — matrix operand order repaired and checked by reading, not compiled, not run on hardware
//
#include <hip/hip_runtime.h>
#include <stdint.h>

typedef __attribute__((ext_vector_type(16))) __bf16 v16bf;
typedef __attribute__((ext_vector_type(8)))  float  v8f;

#define SQ     2048
#define DH     64
#define NH     16
#define NB     2
#define QTILE  16
#define KCHUNK 128
#define NCHUNK (SQ / KCHUNK)

// LDS layout (bytes)
#define OFF_QBF   0        // ushort[16*64]              = 2048
#define OFF_KV    2048     // float[2][128*64] dbl-buf K = 65536 ; phase2: ushort Vt[64][128]
#define OFF_SF    67584    // float [16*2048]            = 131072
#define OFF_PBF   198656   // ushort[16*2048]            = 65536
#define OFF_RED   264192   // float [16*16]              = 1024
#define OFF_ROWM  265216   // float [16]
#define OFF_ROWS  265280   // float [16]
#define OFF_OPART 265344   // float [16*64]              = 4096
#define SMEM_BYTES 269440

__device__ __forceinline__ unsigned short f2bf(float x) {
  union { float f; unsigned u; } c; c.f = x;
  return (unsigned short)((c.u + 0x8000u) >> 16);     // round-to-nearest (ties away)
}
// pack bf16(a) into [15:0], bf16(b) into [31:16] : 2 adds + 1 v_perm_b32
__device__ __forceinline__ unsigned pack2(float a, float b) {
  union { float f; unsigned u; } ca, cb; ca.f = a; cb.f = b;
  return __builtin_amdgcn_perm(cb.u + 0x8000u, ca.u + 0x8000u, 0x07060302u);
}

// async copy: 16 bytes per lane, global -> LDS, tracked by ASYNCcnt
__device__ __forceinline__ void async_ld128(unsigned lds_off, const float* gaddr) {
  asm volatile("global_load_async_to_lds_b128 %0, %1, off"
               :: "v"(lds_off), "v"(gaddr) : "memory");
}

union ABfrag { unsigned u[8]; v16bf v; };

// A matrix 16x32 bf16: lane half h -> K base 16h; VGPR v: K = 16h + 2v
__device__ __forceinline__ v16bf load_A(const unsigned short* base, int row, int stride,
                                        int d0, int half) {
  ABfrag t;
#pragma unroll
  for (int v = 0; v < 8; ++v) {
    int k0 = half * 16 + 2 * v;
    t.u[v] = *(const unsigned*)(base + row * stride + d0 + k0);
  }
  return t.v;
}

// B matrix 32x16 for Q*K^T from fp32 K rows in LDS: B[kd][n] = K[j][d0+kd]
// pairs contiguous -> ds_load_b64 + v_perm pack (co-executes with XDL WMMA)
__device__ __forceinline__ v16bf load_BK32(const float* kb, int jrow, int half, int d0) {
  ABfrag t;
#pragma unroll
  for (int v = 0; v < 8; ++v) {
    int kd = half * 16 + 2 * v;
    const float2 p = *(const float2*)(kb + jrow * DH + d0 + kd);
    t.u[v] = pack2(p.x, p.y);
  }
  return t.v;
}

// B matrix 32x16 for P*V from transposed bf16 Vt[d][k]: pairs contiguous in Vt row
__device__ __forceinline__ v16bf load_BV(const unsigned short* vt, int drow, int kk0, int half) {
  ABfrag t;
#pragma unroll
  for (int v = 0; v < 8; ++v) {
    int k = half * 16 + 2 * v;
    t.u[v] = *(const unsigned*)(vt + drow * KCHUNK + kk0 + k);
  }
  return t.v;
}

__global__ __launch_bounds__(256) void attn_fwd_kernel(
    const float* __restrict__ Q, const float* __restrict__ K,
    const float* __restrict__ V, const int* __restrict__ mask,
    float* __restrict__ Out, float* __restrict__ Wout) {
  extern __shared__ char smem[];
  unsigned short* Qbf = (unsigned short*)(smem + OFF_QBF);
  float*          Sf  = (float*)(smem + OFF_SF);
  unsigned short* Pbf = (unsigned short*)(smem + OFF_PBF);
  float*          red = (float*)(smem + OFF_RED);
  float*          rowm   = (float*)(smem + OFF_ROWM);
  float*          rowsum = (float*)(smem + OFF_ROWS);
  float*          Opart  = (float*)(smem + OFF_OPART);

  const int t    = threadIdx.x;
  const int lane = t & 31, w = t >> 5;
  const int lm   = lane & 15, half = lane >> 4;

  const int head = blockIdx.x >> 7;            // b*NH + h
  const int q0   = (blockIdx.x & 127) * QTILE;
  const int b    = head >> 4;

  const size_t hbase = (size_t)head * SQ * DH;
  const float* Qg = Q + hbase + (size_t)q0 * DH;
  const float* Kg = K + hbase;
  const float* Vg = V + hbase;

  // LDS byte offset of the K double-buffer (flat LDS addr low 32 bits = LDS offset)
  const unsigned kvoff = (unsigned)(uintptr_t)(smem + OFF_KV);

  // ---- prime: async-stage K chunk 0 (fp32, 32 KB) while we stage Q ----
#pragma unroll
  for (int i = 0; i < 8; ++i) {
    int e = (t + 256 * i) * 4;                 // float index (16B granule per lane)
    async_ld128(kvoff + e * 4, Kg + e);
  }

  // ---- stage Q tile (16x64) as bf16 in LDS ----
  for (int i = t; i < QTILE * DH; i += 256) Qbf[i] = f2bf(Qg[i]);
  __syncthreads();

  const v16bf aQ0 = load_A(Qbf, lm, DH, 0, half);
  const v16bf aQ1 = load_A(Qbf, lm, DH, 32, half);

  // ---- phase 1: S = scale * Q K^T, full 16x2048 fp32 row-block in LDS ----
  // K chunks double-buffered via global_load_async_to_lds_b128 / s_wait_asynccnt
  const float scale = 0.125f;                  // 1/sqrt(64)
  for (int kb = 0; kb < NCHUNK; ++kb) {
    if (kb + 1 < NCHUNK) {
      const float* src = Kg + (size_t)(kb + 1) * KCHUNK * DH;
      unsigned dst = kvoff + ((kb + 1) & 1) * 32768u;
#pragma unroll
      for (int i = 0; i < 8; ++i) {
        int e = (t + 256 * i) * 4;
        async_ld128(dst + e * 4, src + e);
      }
      asm volatile("s_wait_asynccnt 0x8" ::: "memory");   // current chunk retired
    } else {
      asm volatile("s_wait_asynccnt 0x0" ::: "memory");
    }
    __syncthreads();

    const float* Kf = (const float*)(smem + OFF_KV + (kb & 1) * 32768);
    v8f acc = {};
    v16bf b0 = load_BK32(Kf, w * 16 + lm, half, 0);
    v16bf b1 = load_BK32(Kf, w * 16 + lm, half, 32);
    acc = __builtin_amdgcn_wmma_f32_16x16x32_bf16(false, aQ0, false, b0, (short)0, acc, false, false);
    acc = __builtin_amdgcn_wmma_f32_16x16x32_bf16(false, aQ1, false, b1, (short)0, acc, false, false);

    const int gcol = kb * KCHUNK + w * 16 + lm;          // D: N = lane&15
#pragma unroll
    for (int v = 0; v < 8; ++v)
      Sf[(half * 8 + v) * SQ + gcol] = acc[v] * scale;   // D: M = 8*half + v
    __syncthreads();
  }

  // ---- phase 1b: mask + row softmax in LDS; write weights (global) + P (bf16, LDS) ----
  // each thread owns 2 adjacent columns -> float2/int2 traffic, packed bf16 stores
  {
    const int r = t >> 4, c2 = (t & 15) * 2;
    const int* mrow = mask + ((size_t)b * SQ + q0 + r) * SQ;
    float mx = -3.0e38f;
    for (int i = 0; i < 64; ++i) {
      int col = c2 + 32 * i;                   // consecutive lanes -> contiguous 8B
      float2 s = *(float2*)&Sf[r * SQ + col];
      int2 mk = *(const int2*)&mrow[col];
      bool wrb = (mk.x <= 0) | (mk.y <= 0);
      if (mk.x <= 0) s.x = -1e9f;
      if (mk.y <= 0) s.y = -1e9f;
      if (wrb) *(float2*)&Sf[r * SQ + col] = s;
      mx = fmaxf(mx, fmaxf(s.x, s.y));
    }
    red[r * 16 + (t & 15)] = mx;
    __syncthreads();
    if ((t & 15) == 0) {
      float m2 = red[r * 16];
      for (int i = 1; i < 16; ++i) m2 = fmaxf(m2, red[r * 16 + i]);
      rowm[r] = m2;
    }
    __syncthreads();
    const float m = rowm[r];
    float sum = 0.f;
    for (int i = 0; i < 64; ++i) {
      int col = c2 + 32 * i;
      float2 s = *(float2*)&Sf[r * SQ + col];
      s.x = __expf(s.x - m);
      s.y = __expf(s.y - m);
      *(float2*)&Sf[r * SQ + col] = s;
      sum += s.x + s.y;
    }
    red[r * 16 + (t & 15)] = sum;
    __syncthreads();
    if ((t & 15) == 0) {
      float s2 = 0.f;
      for (int i = 0; i < 16; ++i) s2 += red[r * 16 + i];
      rowsum[r] = s2;
    }
    __syncthreads();
    const float inv = 1.0f / rowsum[r];
    float* wrow = Wout + ((size_t)head * SQ + q0 + r) * SQ;
    for (int i = 0; i < 64; ++i) {
      int col = c2 + 32 * i;
      float2 p = *(float2*)&Sf[r * SQ + col];
      p.x *= inv; p.y *= inv;
      *(float2*)&wrow[col] = p;                         // 8B coalesced weights store
      *(unsigned*)&Pbf[r * SQ + col] = pack2(p.x, p.y); // 1 dword ds_store per pair
    }
  }
  __syncthreads();

  // ---- phase 2: O = P V, waves split 4 d-tiles x 2 K-halves ----
  unsigned short* Vt = (unsigned short*)(smem + OFF_KV);  // reuse K buffer: Vt[64][128] bf16
  const int dsub = w & 3, khalf = w >> 2;
  v8f oacc = {};
  for (int vb = 0; vb < NCHUNK; ++vb) {
    const float4* src = (const float4*)(Vg + (size_t)vb * KCHUNK * DH);
#pragma unroll
    for (int i = 0; i < 8; ++i) {
      int j = t + 256 * i;
      float4 f = src[j];
      int flat = 4 * j, k = flat >> 6, d = flat & 63;     // transpose into Vt[d][k]
      Vt[(d + 0) * KCHUNK + k] = f2bf(f.x);
      Vt[(d + 1) * KCHUNK + k] = f2bf(f.y);
      Vt[(d + 2) * KCHUNK + k] = f2bf(f.z);
      Vt[(d + 3) * KCHUNK + k] = f2bf(f.w);
    }
    if (vb + 1 < NCHUNK)                       // global_prefetch_b8 next V chunk
      __builtin_prefetch(Vg + (size_t)(vb + 1) * KCHUNK * DH + t * 32, 0, 0);
    __syncthreads();
#pragma unroll
    for (int s = 0; s < 2; ++s) {
      int kk0 = khalf * 64 + s * 32;
      v16bf a  = load_A(Pbf, lm, SQ, vb * KCHUNK + kk0, half);
      v16bf bv = load_BV(Vt, dsub * 16 + lm, kk0, half);
      oacc = __builtin_amdgcn_wmma_f32_16x16x32_bf16(false, a, false, bv, (short)0, oacc, false, false);
    }
    __syncthreads();
  }

  // ---- pairwise reduce wave partials and store O ----
  if (w >= 4) {
#pragma unroll
    for (int v = 0; v < 8; ++v)
      Opart[(half * 8 + v) * DH + dsub * 16 + lm] = oacc[v];
  }
  __syncthreads();
  if (w < 4) {
    float* orow = Out + ((size_t)head * SQ + q0) * DH;
#pragma unroll
    for (int v = 0; v < 8; ++v) {
      int m = half * 8 + v, c = dsub * 16 + lm;
      orow[(size_t)m * DH + c] = oacc[v] + Opart[m * DH + c];
    }
  }
}

extern "C" void kernel_launch(void* const* d_in, const int* in_sizes, int n_in,
                              void* d_out, int out_size, void* d_ws, size_t ws_size,
                              hipStream_t stream) {
  (void)in_sizes; (void)n_in; (void)out_size; (void)d_ws; (void)ws_size;
  const float* Q    = (const float*)d_in[0];
  const float* K    = (const float*)d_in[1];
  const float* V    = (const float*)d_in[2];
  const int*   mask = (const int*)d_in[3];
  float* Out  = (float*)d_out;
  float* Wout = Out + (size_t)NB * NH * SQ * DH;   // weights follow output

  (void)hipFuncSetAttribute((const void*)attn_fwd_kernel,
                            hipFuncAttributeMaxDynamicSharedMemorySize, SMEM_BYTES);

  dim3 grid(NB * NH * (SQ / QTILE));               // 4096 workgroups
  hipLaunchKernelGGL(attn_fwd_kernel, grid, dim3(256), SMEM_BYTES, stream,
                     Q, K, V, mask, Out, Wout);
}
